// refine_ParsingNet_52785148068018
// MI455X (gfx1250) — matrix operand order repaired and checked by reading, not compiled, not run on hardware
//
#include <hip/hip_runtime.h>

typedef _Float16 half16 __attribute__((ext_vector_type(16)));
typedef float    float8 __attribute__((ext_vector_type(8)));

#define EPSN 1e-5f

__device__ __forceinline__ float lrelu_f(float v){ return v >= 0.f ? v : 0.2f*v; }

// ---------------------------------------------------------------------------
// Conv config helpers (K ordering: k = g*16 + cidx, g = tap*CG + cgroup)
// ---------------------------------------------------------------------------
__host__ __device__ constexpr int conv_ceff(int cin, bool coords){ return cin + (coords?2:0); }
__host__ __device__ constexpr int conv_cg(int cin, bool coords){ return (conv_ceff(cin,coords)+15)/16; }
__host__ __device__ constexpr int conv_cpadc(int cin, bool coords){ return conv_cg(cin,coords)*16; }
__host__ __device__ constexpr int conv_g(int cin, bool coords){ return 9*conv_cg(cin,coords); }
__host__ __device__ constexpr int conv_chunks(int cin, bool coords){
  return ((conv_g(cin,coords)+1) & ~1) / 2;
}

// ---------------------------------------------------------------------------
// Sum over channels: out[b,p] = sum_c in[b,c,p]
// ---------------------------------------------------------------------------
__global__ void k_sumc(const float* __restrict__ in, float* __restrict__ out,
                       int B, int C, int HW){
  int idx = blockIdx.x*blockDim.x + threadIdx.x;
  if (idx >= B*HW) return;
  int b = idx / HW, p = idx % HW;
  const float* xb = in + (size_t)b*C*HW + p;
  float s = 0.f;
  for (int c = 0; c < C; ++c) s += xb[(size_t)c*HW];
  out[idx] = s;
}

// ---------------------------------------------------------------------------
// share_block: conv3x3(cin=1|2) -> InstanceNorm(128x128) -> LeakyReLU -> down2
// ---------------------------------------------------------------------------
__device__ __forceinline__ float share_conv_at(const float* __restrict__ x0,
                                               const float* __restrict__ x1,
                                               const float* __restrict__ w,
                                               float bias0, int hasUp, int y, int x){
  float acc = bias0;
  #pragma unroll
  for (int ky = 0; ky < 3; ++ky){
    #pragma unroll
    for (int kx = 0; kx < 3; ++kx){
      int yy = y + ky - 1, xx = x + kx - 1;
      if ((unsigned)yy < 128u && (unsigned)xx < 128u){
        acc += w[ky*3+kx] * x0[yy*128 + xx];
        if (hasUp) acc += w[9 + ky*3 + kx] * x1[(yy>>1)*64 + (xx>>1)];
      }
    }
  }
  return acc;
}

__global__ __launch_bounds__(1024)
void k_share_block(const float* __restrict__ in0, const float* __restrict__ in1,
                   const float* __restrict__ w, const float* __restrict__ bias,
                   float* __restrict__ out, int hasUp){
  __shared__ float rs[1024], rs2[1024];
  int b = blockIdx.x;
  const float* x0 = in0 + (size_t)b*16384;
  const float* x1 = hasUp ? (in1 + (size_t)b*4096) : nullptr;
  float b0 = bias[0];
  float s = 0.f, s2 = 0.f;
  for (int p = threadIdx.x; p < 16384; p += 1024){
    float acc = share_conv_at(x0, x1, w, b0, hasUp, p >> 7, p & 127);
    s += acc; s2 += acc*acc;
  }
  rs[threadIdx.x] = s; rs2[threadIdx.x] = s2;
  __syncthreads();
  for (int o = 512; o > 0; o >>= 1){
    if ((int)threadIdx.x < o){ rs[threadIdx.x] += rs[threadIdx.x+o]; rs2[threadIdx.x] += rs2[threadIdx.x+o]; }
    __syncthreads();
  }
  float mu  = rs[0]  * (1.f/16384.f);
  float var = rs2[0] * (1.f/16384.f) - mu*mu;
  float inv = rsqrtf(var + EPSN);
  for (int q = threadIdx.x; q < 4096; q += 1024){
    int y = (q >> 6) << 1, x = (q & 63) << 1;   // down2: even rows/cols
    float acc = share_conv_at(x0, x1, w, b0, hasUp, y, x);
    out[(size_t)b*4096 + q] = lrelu_f((acc - mu) * inv);
  }
}

// ---------------------------------------------------------------------------
// InstanceNorm + LeakyReLU, one block per (b,c). In-place safe.
// ---------------------------------------------------------------------------
__global__ __launch_bounds__(256)
void k_in_lrelu(const float* __restrict__ in, float* __restrict__ out, int HW){
  int bc = blockIdx.x;
  const float* x = in  + (size_t)bc*HW;
  float*       y = out + (size_t)bc*HW;
  float s = 0.f, s2 = 0.f;
  for (int i = threadIdx.x; i < HW; i += 256){ float v = x[i]; s += v; s2 += v*v; }
  __shared__ float rs[256], rs2[256];
  rs[threadIdx.x] = s; rs2[threadIdx.x] = s2;
  __syncthreads();
  for (int o = 128; o > 0; o >>= 1){
    if ((int)threadIdx.x < o){ rs[threadIdx.x] += rs[threadIdx.x+o]; rs2[threadIdx.x] += rs2[threadIdx.x+o]; }
    __syncthreads();
  }
  float mu  = rs[0]  / (float)HW;
  float var = rs2[0] / (float)HW - mu*mu;
  float inv = rsqrtf(var + EPSN);
  for (int i = threadIdx.x; i < HW; i += 256) y[i] = lrelu_f((x[i] - mu) * inv);
}

// ---------------------------------------------------------------------------
// Input prep: fp32 NCHW [B,CIN,H,W] -> zero-padded fp16 NHWC [B,H+2,W+2,CPADC]
// (borders zero, coord channels baked in, channels zero-padded to 16*CG)
// ---------------------------------------------------------------------------
template<int CIN, bool COORDS, int H, int W>
__global__ void k_prep_input(const float* __restrict__ x, _Float16* __restrict__ dst, int B){
  constexpr int CPADC = conv_cpadc(CIN, COORDS);
  constexpr int PH = H+2, PW = W+2;
  int idx = blockIdx.x*blockDim.x + threadIdx.x;
  if (idx >= B*PH*PW*CPADC) return;
  int c  = idx % CPADC;
  int px = (idx / CPADC) % PW;
  int py = (idx / (CPADC*PW)) % PH;
  int b  = idx / (CPADC*PW*PH);
  int xx = px - 1, yy = py - 1;
  float v = 0.f;
  if ((unsigned)yy < (unsigned)H && (unsigned)xx < (unsigned)W){
    if (c < CIN)                       v = x[(((size_t)b*CIN + c)*H + yy)*W + xx];
    else if (COORDS && c == CIN)       v = (float)xx * (2.f/(float)(W-1)) - 1.f;
    else if (COORDS && c == CIN + 1)   v = (float)yy * (2.f/(float)(H-1)) - 1.f;
  }
  dst[idx] = (_Float16)v;
}

// ---------------------------------------------------------------------------
// Weight prep: fp32 [COUT][CEFF][3][3] -> fp16 A-fragment layout
// [MT][CHUNK][lane][16] using the tap-group K ordering:
//   k = g*16 + cidx,  g = tap*CG + cgroup,  channel = cgroup*16 + cidx
// lane/element map: k = ch*32 + 16*(lane/16) + j (contiguous k order, same as B).
// ---------------------------------------------------------------------------
__global__ void k_prep_wfrag(const float* __restrict__ w, _Float16* __restrict__ wf,
                             int CEFF, int CG, int COUT, int MT){
  int G = 9*CG;
  int chunks = ((G+1) & ~1) / 2;
  int total = MT*chunks*32*16;
  int idx = blockIdx.x*blockDim.x + threadIdx.x;
  if (idx >= total) return;
  int j    = idx & 15;
  int lane = (idx >> 4) & 31;
  int ch   = (idx >> 9) % chunks;
  int mt   = idx / (chunks*512);
  int k = ch*32 + (lane & 16) + j;
  int g = k >> 4, cidx = k & 15;
  int pos = g / CG, cgi = g % CG;          // pos = ky*3+kx
  int channel = cgi*16 + cidx;
  int m = mt*16 + (lane & 15);
  float v = 0.f;
  if (g < G && channel < CEFF && m < COUT)
    v = w[((size_t)m*CEFF + channel)*9 + pos];
  wf[idx] = (_Float16)v;
}

// ---------------------------------------------------------------------------
// 3x3 conv as implicit GEMM on WMMA f32_16x16x32_f16, fully vector fragments:
//   A: pre-swizzled fp16 weight fragments -> 2x global_load_b128 per lane/chunk
//   B: NHWC fp16 image -> 16 consecutive channels at one tap pixel
//      = ONE 32-byte half16 load (2x b128); tap offsets fold to immediates,
//      lane-half select is a single cndmask. No scalar gathers, no branches.
// One wave per (b, y, 16-wide x strip); wave-uniform control flow.
// ---------------------------------------------------------------------------
template<int CIN, int COUT, bool COORDS, int H, int W>
__global__ __launch_bounds__(256)
void k_conv3x3_wmma(const _Float16* __restrict__ inPad, const _Float16* __restrict__ wf,
                    const float* __restrict__ bias, float* __restrict__ out, int B){
  constexpr int CG     = conv_cg(CIN, COORDS);
  constexpr int CPADC  = conv_cpadc(CIN, COORDS);
  constexpr int G      = conv_g(CIN, COORDS);
  constexpr int CHUNKS = conv_chunks(CIN, COORDS);
  constexpr int PH = H+2, PW = W+2;
  constexpr int NXT = W/16;

  const int lane = threadIdx.x & 31;
  const int wid  = threadIdx.x >> 5;
  int tile  = blockIdx.x*8 + wid;
  int total = B * H * NXT;
  if (tile >= total) return;                 // wave-uniform exit

  const int xt = tile % NXT;
  const int y  = (tile / NXT) % H;
  const int b  = tile / (NXT * H);
  const int x0 = xt * 16;
  const int xB = x0 + (lane & 15);
  const bool hiLane = (lane & 16) != 0;

  // base at (b, y, xB, c=0); tap offsets are added as compile-time constants
  const _Float16* __restrict__ src =
      inPad + ((size_t)b*PH*PW + (size_t)y*PW + xB) * CPADC;
  const half16*  __restrict__ wfv = (const half16*)wf;   // [mt][chunk][lane]

  float8 acc0 = {}; float8 acc1 = {};

  #pragma unroll
  for (int ch = 0; ch < CHUNKS; ++ch){
    half16 a0 = wfv[(size_t)ch*32 + lane];
    half16 a1v = {};
    if constexpr (COUT > 16) a1v = wfv[(size_t)(CHUNKS + ch)*32 + lane];

    const int gLo = 2*ch;
    const int gHi = (2*ch+1 < G) ? (2*ch+1) : (G-1);   // padded group -> A is zero
    const int posLo = gLo / CG, posHi = gHi / CG;
    const int oLo = ((posLo/3)*PW + (posLo%3))*CPADC + (gLo%CG)*16;  // immediate
    const int oHi = ((posHi/3)*PW + (posHi%3))*CPADC + (gHi%CG)*16;  // immediate
    half16 bf = *(const half16*)(src + (hiLane ? oHi : oLo));

    acc0 = __builtin_amdgcn_wmma_f32_16x16x32_f16(false, a0, false, bf, (short)0, acc0, false, false);
    if constexpr (COUT > 16)
      acc1 = __builtin_amdgcn_wmma_f32_16x16x32_f16(false, a1v, false, bf, (short)0, acc1, false, false);
  }

  // store D (f32 C/D: VGPR r, lanes 0-15 -> M=r, lanes 16-31 -> M=r+8)
  const int n   = lane & 15;
  const int mhi = hiLane ? 8 : 0;
  #pragma unroll
  for (int r = 0; r < 8; ++r){
    int m = r + mhi;
    if (m < COUT)
      out[(((size_t)b*COUT + m)*H + y)*W + x0 + n] = acc0[r] + bias[m];
    if constexpr (COUT > 16)
      out[(((size_t)b*COUT + (m+16))*H + y)*W + x0 + n] = acc1[r] + bias[m+16];
  }
}

// ---------------------------------------------------------------------------
// Learnable shortcut: out += conv1x1(add_coords(x))
// ---------------------------------------------------------------------------
__global__ void k_shortcut_add(const float* __restrict__ x, const float* __restrict__ w,
                               const float* __restrict__ b, float* __restrict__ out,
                               int B, int cin, int cout, int H, int W){
  int HW = H*W;
  int idx = blockIdx.x*blockDim.x + threadIdx.x;
  if (idx >= B*cout*HW) return;
  int p  = idx % HW;
  int co = (idx / HW) % cout;
  int bb = idx / (HW*cout);
  int y = p / W, xx = p % W;
  const float* wr = w + (size_t)co*(cin+2);
  const float* xb = x + (size_t)bb*cin*HW;
  float acc = b[co];
  for (int c = 0; c < cin; ++c) acc += wr[c] * xb[(size_t)c*HW + p];
  acc += wr[cin]   * ((float)xx * (2.f/(float)(W-1)) - 1.f);
  acc += wr[cin+1] * ((float)y  * (2.f/(float)(H-1)) - 1.f);
  out[idx] += acc;
}

// down2 (nearest, stride 2): [BC,128,128] -> [BC,64,64]
__global__ void k_down2(const float* __restrict__ in, float* __restrict__ out, int BC){
  int idx = blockIdx.x*blockDim.x + threadIdx.x;
  if (idx >= BC*4096) return;
  int q  = idx % 4096;
  int bc = idx / 4096;
  int y = q >> 6, x = q & 63;
  out[idx] = in[(size_t)bc*16384 + (y*2)*128 + x*2];
}

// Rank-1 contraction: s[b,c] = sum_n mat[b,c,n] * vec[b,n]
__global__ __launch_bounds__(256)
void k_rowdot(const float* __restrict__ mat, const float* __restrict__ vec,
              float* __restrict__ s, int C, int hw){
  int bc = blockIdx.x;
  int b  = bc / C;
  const float* row = mat + (size_t)bc*hw;
  const float* vv  = vec + (size_t)b*hw;
  float acc = 0.f;
  for (int i = threadIdx.x; i < hw; i += 256) acc += row[i]*vv[i];
  __shared__ float rs[256];
  rs[threadIdx.x] = acc;
  __syncthreads();
  for (int o = 128; o > 0; o >>= 1){
    if ((int)threadIdx.x < o) rs[threadIdx.x] += rs[threadIdx.x+o];
    __syncthreads();
  }
  if (threadIdx.x == 0) s[bc] = rs[0];
}

// out_0 = up2( s0[b,c] * a2_0[b,m] * mask_0[b,m] )
__global__ void k_out0(const float* __restrict__ s, const float* __restrict__ a2,
                       const float* __restrict__ mask, float* __restrict__ out, int B, int C){
  int idx = blockIdx.x*blockDim.x + threadIdx.x;
  if (idx >= B*C*16384) return;
  int p = idx % 16384;
  int c = (idx / 16384) % C;
  int b = idx / (16384*C);
  int Y = p >> 7, X = p & 127;
  int m = (Y>>1)*64 + (X>>1);
  out[idx] = s[b*C + c] * a2[b*4096 + m] * mask[b*4096 + m];
}

// out_1 = up2( (s1[b,c]*a2_1[b,m] + bias1[b,c,m]) * mask_1[b,m] )
__global__ void k_out1(const float* __restrict__ s, const float* __restrict__ a2,
                       const float* __restrict__ bias1, const float* __restrict__ mask,
                       float* __restrict__ out, int B, int C){
  int idx = blockIdx.x*blockDim.x + threadIdx.x;
  if (idx >= B*C*16384) return;
  int p = idx % 16384;
  int c = (idx / 16384) % C;
  int b = idx / (16384*C);
  int Y = p >> 7, X = p & 127;
  int m = (Y>>1)*64 + (X>>1);
  out[idx] = (s[b*C + c] * a2[b*4096 + m] + bias1[((size_t)(b*C + c))*4096 + m]) * mask[b*4096 + m];
}

__global__ void k_avg(const float* __restrict__ p0, const float* __restrict__ p1,
                      float* __restrict__ out, int n){
  int idx = blockIdx.x*blockDim.x + threadIdx.x;
  if (idx < n) out[idx] = 0.5f * (p0[idx] + p1[idx]);
}

// ---------------------------------------------------------------------------
// Host
// ---------------------------------------------------------------------------
extern "C" void kernel_launch(void* const* d_in, const int* in_sizes, int n_in,
                              void* d_out, int out_size, void* d_ws, size_t ws_size,
                              hipStream_t stream){
  const int B = 2, CA = 8, Hc = 128, Wc = 128, HW = Hc*Wc, hw = 64*64;

  const float* a1 = (const float*)d_in[0];
  const float* b1 = (const float*)d_in[1];
  const float* a2 = (const float*)d_in[2];

  // --- parameter pointer resolution (dict-insertion order vs jax-sorted pytree) ---
  const float* pp[28];
  for (int i = 0; i < 28 && 3+i < n_in; ++i) pp[i] = (const float*)d_in[3+i];
  const float *cs1_w,*cs1_b,*cs2_w,*cs2_b,*cm1_w,*cm1_b,*cm2_w,*cm2_b;
  const float *rb1_w1,*rb1_b1,*rb1_w2,*rb1_b2,*rb1_ws,*rb1_bs;
  const float *rb2_w1,*rb2_b1,*rb2_w2,*rb2_b2,*rb2_ws,*rb2_bs;
  const float *rbc_w1,*rbc_b1,*rbc_w2,*rbc_b2,*rbc_ws,*rbc_bs;
  const float *po_w,*po_b;
  if (in_sizes[3] == 9){
    cs1_w=pp[0]; cs1_b=pp[1]; cs2_w=pp[2]; cs2_b=pp[3];
    cm1_w=pp[4]; cm1_b=pp[5]; cm2_w=pp[6]; cm2_b=pp[7];
    rb1_w1=pp[8];  rb1_b1=pp[9];  rb1_w2=pp[10]; rb1_b2=pp[11]; rb1_ws=pp[12]; rb1_bs=pp[13];
    rb2_w1=pp[14]; rb2_b1=pp[15]; rb2_w2=pp[16]; rb2_b2=pp[17]; rb2_ws=pp[18]; rb2_bs=pp[19];
    rbc_w1=pp[20]; rbc_b1=pp[21]; rbc_w2=pp[22]; rbc_b2=pp[23]; rbc_ws=pp[24]; rbc_bs=pp[25];
    po_w=pp[26]; po_b=pp[27];
  } else {
    rb1_b1=pp[0];  rb1_b2=pp[1];  rb1_bs=pp[2];  rb1_w1=pp[3];  rb1_w2=pp[4];  rb1_ws=pp[5];
    rb2_b1=pp[6];  rb2_b2=pp[7];  rb2_bs=pp[8];  rb2_w1=pp[9];  rb2_w2=pp[10]; rb2_ws=pp[11];
    rbc_b1=pp[12]; rbc_b2=pp[13]; rbc_bs=pp[14]; rbc_w1=pp[15]; rbc_w2=pp[16]; rbc_ws=pp[17];
    cm1_b=pp[18]; cm1_w=pp[19]; cm2_b=pp[20]; cm2_w=pp[21];
    cs1_b=pp[22]; cs1_w=pp[23]; cs2_b=pp[24]; cs2_w=pp[25];
    po_b=pp[26]; po_w=pp[27];
  }

  // --- workspace carve-up ---
  float* Wp = (float*)d_ws;
  size_t off = 0;
  auto alloc = [&](size_t n){ float* p = Wp + off; off += (n + 15) & ~(size_t)15; return p; };
  float* A1S   = alloc(B*HW);
  float* A2S   = alloc(B*HW);
  float* A10   = alloc(B*hw);
  float* A20   = alloc(B*hw);
  float* M0    = alloc(B*hw);
  float* A11   = alloc(B*hw);
  float* A21   = alloc(B*hw);
  float* M1    = alloc(B*hw);
  float* B10   = alloc((size_t)B*32*hw);
  float* B11   = alloc((size_t)B*32*hw);
  float* BIAS1 = alloc((size_t)B*32*hw);
  float* S0    = alloc(64);
  float* S1    = alloc(64);
  float* XN    = alloc((size_t)B*32*HW);
  float* T1    = alloc((size_t)B*32*HW);
  float* T2    = alloc((size_t)B*32*HW);
  float* OUT0  = alloc((size_t)B*32*HW);
  float* P0    = alloc((size_t)B*8*HW);
  float* P1    = alloc((size_t)B*8*HW);
  // fp16 staging: NHWC padded image (max 2*130*130*48 halves) + weight fragments
  _Float16* INPAD = (_Float16*)alloc(812096);        // 1,624,192 halves capacity
  _Float16* WFA   = (_Float16*)alloc(41024);         // 82,048 halves capacity
  float* OUT1  = T1;                                 // reuse
  if (ws_size < off * sizeof(float)) return;

  // weight-fragment sub-allocations (halves; all offsets multiple of 16 -> 32B aligned)
  _Float16* WF_rb1w1 = WFA + 0;       // MT=2, CHUNKS=5  -> 5120
  _Float16* WF_rb1w2 = WFA + 5120;    // MT=2, CHUNKS=14 -> 14336
  _Float16* WF_rb2w1 = WFA + 19456;   // 14336
  _Float16* WF_rb2w2 = WFA + 33792;   // 14336
  _Float16* WF_rbcw1 = WFA + 48128;   // 14336
  _Float16* WF_rbcw2 = WFA + 62464;   // 14336
  _Float16* WF_po    = WFA + 76800;   // MT=1, CHUNKS=9  -> 4608

  auto cdiv = [](int a, int b){ return (a + b - 1)/b; };

  // 0) pre-swizzle all conv weights into A-fragment layout (tap-group K order)
  k_prep_wfrag<<<cdiv(2*5*512,256),256,0,stream>>>(rb1_w1, WF_rb1w1, 10, 1, 32, 2);
  k_prep_wfrag<<<cdiv(2*14*512,256),256,0,stream>>>(rb1_w2, WF_rb1w2, 34, 3, 32, 2);
  k_prep_wfrag<<<cdiv(2*14*512,256),256,0,stream>>>(rb2_w1, WF_rb2w1, 34, 3, 32, 2);
  k_prep_wfrag<<<cdiv(2*14*512,256),256,0,stream>>>(rb2_w2, WF_rb2w2, 34, 3, 32, 2);
  k_prep_wfrag<<<cdiv(2*14*512,256),256,0,stream>>>(rbc_w1, WF_rbcw1, 34, 3, 32, 2);
  k_prep_wfrag<<<cdiv(2*14*512,256),256,0,stream>>>(rbc_w2, WF_rbcw2, 34, 3, 32, 2);
  k_prep_wfrag<<<cdiv(1*9*512,256),256,0,stream>>>(po_w,   WF_po,    32, 2,  8, 1);

  constexpr int NP10_128 = 2*130*130*conv_cpadc(8,true);    // 16 ch
  constexpr int NP34_128 = 2*130*130*conv_cpadc(32,true);   // 48 ch
  constexpr int NP34_64  = 2*66*66*conv_cpadc(32,true);
  constexpr int NP32_128 = 2*130*130*conv_cpadc(32,false);  // 32 ch
  int tiles128 = B*Hc*(Wc/16), tiles64 = B*64*(64/16);

  // 1) channel sums
  k_sumc<<<cdiv(B*HW,256),256,0,stream>>>(a1, A1S, B, CA, HW);
  k_sumc<<<cdiv(B*HW,256),256,0,stream>>>(a2, A2S, B, CA, HW);

  // 2) level-0 share blocks
  k_share_block<<<B,1024,0,stream>>>(A1S, nullptr, cs1_w, cs1_b, A10, 0);
  k_share_block<<<B,1024,0,stream>>>(A2S, nullptr, cs1_w, cs1_b, A20, 0);
  k_share_block<<<B,1024,0,stream>>>(A2S, nullptr, cm1_w, cm1_b, M0, 0);

  // 3) resblock conv_b_1 on b1 [2,8,128,128] -> T2; b1_0 = down2
  k_in_lrelu<<<B*CA,256,0,stream>>>(b1, XN, HW);
  k_prep_input<8,true,128,128><<<cdiv(NP10_128,256),256,0,stream>>>(XN, INPAD, B);
  k_conv3x3_wmma<8,32,true,128,128><<<cdiv(tiles128,8),256,0,stream>>>(INPAD, WF_rb1w1, rb1_b1, T1, B);
  k_in_lrelu<<<B*32,256,0,stream>>>(T1, T1, HW);
  k_prep_input<32,true,128,128><<<cdiv(NP34_128,256),256,0,stream>>>(T1, INPAD, B);
  k_conv3x3_wmma<32,32,true,128,128><<<cdiv(tiles128,8),256,0,stream>>>(INPAD, WF_rb1w2, rb1_b2, T2, B);
  k_shortcut_add<<<cdiv(B*32*HW,256),256,0,stream>>>(b1, rb1_ws, rb1_bs, T2, B, 8, 32, Hc, Wc);
  k_down2<<<cdiv(B*32*hw,256),256,0,stream>>>(T2, B10, B*32);

  // 4) rank-1 affinity 0
  k_rowdot<<<B*32,256,0,stream>>>(B10, A10, S0, 32, hw);
  k_out0<<<cdiv(B*32*HW,256),256,0,stream>>>(S0, A20, M0, OUT0, B, 32);

  // 5) level-1 share blocks
  k_share_block<<<B,1024,0,stream>>>(A1S, A10, cs2_w, cs2_b, A11, 1);
  k_share_block<<<B,1024,0,stream>>>(A2S, A20, cs2_w, cs2_b, A21, 1);
  k_share_block<<<B,1024,0,stream>>>(A2S, M0,  cm2_w, cm2_b, M1,  1);

  // 6) resblock conv_b_2 on out_0 -> T2; b1_1 = down2
  k_in_lrelu<<<B*32,256,0,stream>>>(OUT0, XN, HW);
  k_prep_input<32,true,128,128><<<cdiv(NP34_128,256),256,0,stream>>>(XN, INPAD, B);
  k_conv3x3_wmma<32,32,true,128,128><<<cdiv(tiles128,8),256,0,stream>>>(INPAD, WF_rb2w1, rb2_b1, T1, B);
  k_in_lrelu<<<B*32,256,0,stream>>>(T1, T1, HW);
  k_prep_input<32,true,128,128><<<cdiv(NP34_128,256),256,0,stream>>>(T1, INPAD, B);
  k_conv3x3_wmma<32,32,true,128,128><<<cdiv(tiles128,8),256,0,stream>>>(INPAD, WF_rb2w2, rb2_b2, T2, B);
  k_shortcut_add<<<cdiv(B*32*HW,256),256,0,stream>>>(OUT0, rb2_ws, rb2_bs, T2, B, 32, 32, Hc, Wc);
  k_down2<<<cdiv(B*32*hw,256),256,0,stream>>>(T2, B11, B*32);

  // 7) resblock conv_b_cut_1 on b1_0 (64x64) -> bias_1
  k_in_lrelu<<<B*32,256,0,stream>>>(B10, XN, hw);
  k_prep_input<32,true,64,64><<<cdiv(NP34_64,256),256,0,stream>>>(XN, INPAD, B);
  k_conv3x3_wmma<32,32,true,64,64><<<cdiv(tiles64,8),256,0,stream>>>(INPAD, WF_rbcw1, rbc_b1, T1, B);
  k_in_lrelu<<<B*32,256,0,stream>>>(T1, T1, hw);
  k_prep_input<32,true,64,64><<<cdiv(NP34_64,256),256,0,stream>>>(T1, INPAD, B);
  k_conv3x3_wmma<32,32,true,64,64><<<cdiv(tiles64,8),256,0,stream>>>(INPAD, WF_rbcw2, rbc_b2, BIAS1, B);
  k_shortcut_add<<<cdiv(B*32*hw,256),256,0,stream>>>(B10, rbc_ws, rbc_bs, BIAS1, B, 32, 32, 64, 64);

  // 8) rank-1 affinity 1 (OUT1 aliases T1)
  k_rowdot<<<B*32,256,0,stream>>>(B11, A11, S1, 32, hw);
  k_out1<<<cdiv(B*32*HW,256),256,0,stream>>>(S1, A21, BIAS1, M1, OUT1, B, 32);

  // 9) parout on both branches, then average
  k_prep_input<32,false,128,128><<<cdiv(NP32_128,256),256,0,stream>>>(OUT0, INPAD, B);
  k_conv3x3_wmma<32,8,false,128,128><<<cdiv(tiles128,8),256,0,stream>>>(INPAD, WF_po, po_b, P0, B);
  k_in_lrelu<<<B*8,256,0,stream>>>(P0, P0, HW);
  k_prep_input<32,false,128,128><<<cdiv(NP32_128,256),256,0,stream>>>(OUT1, INPAD, B);
  k_conv3x3_wmma<32,8,false,128,128><<<cdiv(tiles128,8),256,0,stream>>>(INPAD, WF_po, po_b, P1, B);
  k_in_lrelu<<<B*8,256,0,stream>>>(P1, P1, HW);
  k_avg<<<cdiv(B*8*HW,256),256,0,stream>>>(P0, P1, (float*)d_out, B*8*HW);
}
